// IPA_25168508354962
// MI455X (gfx1250) — compile-verified, matrix operand order checked by reading, not run on hardware
//
#include <hip/hip_runtime.h>
#include <hip/hip_bf16.h>
#include <math.h>
#include <stdint.h>

// ---------------------------------------------------------------------------
// AF2 IPA structure module for MI455X (gfx1250, wave32, WMMA + TDM).
// Big GEMMs: TDM (tensor_load_to_lds) double-buffered LDS panels, 8 waves do
// 2x4 register-blocked v_wmma_f32_16x16x32_bf16 from LDS.
// Odd-width GEMMs: direct-from-global WMMA. LN/softmax/points: fp32 VALU.
// ---------------------------------------------------------------------------

typedef __attribute__((ext_vector_type(16))) __bf16 v16bf;
typedef __attribute__((ext_vector_type(8)))  __bf16 v8bf;
typedef __attribute__((ext_vector_type(8)))  float  v8f;
typedef __attribute__((ext_vector_type(4)))  unsigned v4u;
typedef __attribute__((ext_vector_type(8)))  int    v8i;
typedef __attribute__((ext_vector_type(4)))  int    v4i;

#define FLAG_RELU 1
#define FLAG_ACC  2

#if __has_builtin(__builtin_amdgcn_tensor_load_to_lds) && \
    __has_builtin(__builtin_amdgcn_s_wait_tensorcnt)
#define HAVE_TDM 1
#else
#define HAVE_TDM 0
#endif

// Per-lane bf16 fragment load per CDNA5 ISA layout:
// lane r (=lane&15) holds row r; K-halves [kb..kb+7] and [kb+16..kb+23].
static __device__ __forceinline__ v16bf load_frag32(const __bf16* p) {
  v8bf lo = *(const v8bf*)(p);
  v8bf hi = *(const v8bf*)(p + 16);
  v16bf r;
#pragma unroll
  for (int t = 0; t < 8; ++t) { r[t] = lo[t]; r[8 + t] = hi[t]; }
  return r;
}

static __device__ __forceinline__ v8f wmma_bf16(v16bf a, v16bf b, v8f c) {
  return __builtin_amdgcn_wmma_f32_16x16x32_bf16(false, a, false, b, (short)0, c,
                                                 false, false);
}

#if HAVE_TDM
// Issue one 2D TDM load: tile_rows x tile_k bf16 elements, row stride
// `stride` elements in global memory, packed contiguously into LDS at lds_off.
// D# built per CDNA5 ISA ch.8 (group0 128b + group1 256b; groups 2/3 zero).
// This toolchain exposes the 6-arg builtin: (g0, g1, g2, g3, g4, cpol).
static __device__ __forceinline__ void tdm_load_2d(unsigned lds_off,
                                                   const void* gaddr,
                                                   unsigned tile_k,
                                                   unsigned tile_rows,
                                                   long stride) {
  unsigned long long ga = (unsigned long long)(uintptr_t)gaddr;
  unsigned long long st = (unsigned long long)stride;
  v4u g0;
  g0[0] = 1u;                                   // count=1, user descriptor
  g0[1] = lds_off;                              // lds_addr (bytes)
  g0[2] = (unsigned)(ga & 0xFFFFFFFFu);         // global_addr[31:0]
  g0[3] = (unsigned)((ga >> 32) & 0x01FFFFFFu)  // global_addr[56:32]
        | (2u << 30);                           // type = 2 ("image")
  v8i g1;
  g1[0] = (int)(1u << 16);                      // data_size=1 (2 bytes)
  g1[1] = (int)((tile_k & 0xFFFFu) << 16);      // tensor_dim0[15:0] @ bits63:48
  g1[2] = (int)((tile_k >> 16) | ((tile_rows & 0xFFFFu) << 16));  // td0 hi|td1 lo
  g1[3] = (int)((tile_rows >> 16) | (tile_k << 16));              // td1 hi|tile_dim0
  g1[4] = (int)(tile_rows & 0xFFFFu);           // tile_dim1 (tile_dim2=0)
  g1[5] = (int)(st & 0xFFFFFFFFu);              // tensor_dim0_stride[31:0]
  g1[6] = (int)((st >> 32) & 0xFFFFu);          // stride[47:32] (dim1 stride=0)
  g1[7] = 0;
  v4i z4 = {0, 0, 0, 0};
  v8i z8 = {0, 0, 0, 0, 0, 0, 0, 0};
  __builtin_amdgcn_tensor_load_to_lds(g0, g1, z4, z4, z8, 0);
}
#endif

// --------------------------- f32 -> bf16 convert ---------------------------
__global__ void cvt_kernel(const float* __restrict__ in, __bf16* __restrict__ out,
                           int n) {
  int i = blockIdx.x * 256 + threadIdx.x;
  if (i < n) out[i] = (__bf16)in[i];
}

// zero the padded head rows (h = 12..15) of the attention-prob buffer
__global__ void zero_pad_a_kernel(__bf16* __restrict__ a) {
  int idx = blockIdx.x * 256 + threadIdx.x;  // 384 * 4 * 384 = 589824
  if (idx >= 589824) return;
  int i = idx / 1536;
  int rest = idx % 1536;
  a[((long)i * 16 + 12) * 384 + rest] = (__bf16)0.0f;
}

// ------------------------------- LayerNorm ---------------------------------
__global__ void __launch_bounds__(256)
ln_kernel(const float* __restrict__ x, const float* __restrict__ g,
          const float* __restrict__ bb, float* __restrict__ yf,
          __bf16* __restrict__ yb, int cols) {
  long row = blockIdx.x;
  const float* xr = x + row * cols;
  int t = threadIdx.x;
  int c1 = t + 256;
  float x0 = (t < cols) ? xr[t] : 0.f;
  float x1 = (c1 < cols) ? xr[c1] : 0.f;
  float s = x0 + x1, ss = x0 * x0 + x1 * x1;
#pragma unroll
  for (int off = 16; off; off >>= 1) {
    s += __shfl_xor(s, off, 32);
    ss += __shfl_xor(ss, off, 32);
  }
  __shared__ float sm[16];
  int wv = t >> 5;
  if ((t & 31) == 0) { sm[wv] = s; sm[8 + wv] = ss; }
  __syncthreads();
  if (t == 0) {
    float a = 0.f, b2 = 0.f;
    for (int w = 0; w < 8; ++w) { a += sm[w]; b2 += sm[8 + w]; }
    sm[0] = a; sm[8] = b2;
  }
  __syncthreads();
  float mean = sm[0] / (float)cols;
  float var = sm[8] / (float)cols - mean * mean;
  float inv = rsqrtf(var + 1e-5f);
  if (t < cols) {
    float v = (x0 - mean) * inv * g[t] + bb[t];
    if (yf) yf[row * cols + t] = v;
    if (yb) yb[row * cols + t] = (__bf16)v;
  }
  if (c1 < cols) {
    float v = (x1 - mean) * inv * g[c1] + bb[c1];
    if (yf) yf[row * cols + c1] = v;
    if (yb) yb[row * cols + c1] = (__bf16)v;
  }
}

// ------------------- generic GEMM (odd tile widths) ------------------------
__global__ void __launch_bounds__(256)
gemm_bf16_kernel(const __bf16* __restrict__ A, long lda,
                 const __bf16* __restrict__ W, long ldw,
                 const float* __restrict__ bias, float* __restrict__ Cf, long ldc,
                 __bf16* __restrict__ Cb, long ldcb, int M, int Nc, int K,
                 int flags, int tiles) {
  int wave = threadIdx.x >> 5;
  int lane = threadIdx.x & 31;
  int tile = blockIdx.x * 8 + wave;
  if (tile >= tiles) return;
  int ntn = Nc >> 4;
  int tm = tile / ntn, tn = tile % ntn;
  int r = lane & 15;
  int kh = (lane & 16) ? 8 : 0;
  const __bf16* Ap = A + (long)(tm * 16 + r) * lda + kh;
  const __bf16* Wp = W + (long)(tn * 16 + r) * ldw + kh;
  v8f acc = {0.f, 0.f, 0.f, 0.f, 0.f, 0.f, 0.f, 0.f};
  for (int k = 0; k < K; k += 32) {
    __builtin_prefetch(Ap + k + 256, 0, 1);  // global_prefetch_b8
    __builtin_prefetch(Wp + k + 256, 0, 1);
    acc = wmma_bf16(load_frag32(Ap + k), load_frag32(Wp + k), acc);
  }
  int col = tn * 16 + r;
  int m0 = (lane & 16) ? 8 : 0;
  float bv = bias ? bias[col] : 0.f;
#pragma unroll
  for (int t = 0; t < 8; ++t) {
    long row = tm * 16 + m0 + t;
    float v = acc[t] + bv;
    if (flags & FLAG_ACC) v += Cf[row * ldc + col];
    if (flags & FLAG_RELU) v = fmaxf(v, 0.f);
    if (Cf) Cf[row * ldc + col] = v;
    if (Cb) Cb[row * ldcb + col] = (__bf16)v;
  }
}

// ---------------- TDM GEMM: M,Nc multiples of 128, K of 64 -----------------
// Block: 128x128 output macro-tile. Wave w: 32x64 (2x4 16x16 subtiles).
// Wave 0 issues TDM panel loads (A 128x64, W 128x64) double-buffered in LDS.
__global__ void __launch_bounds__(256)
gemm_tdm_kernel(const __bf16* __restrict__ A, long lda,
                const __bf16* __restrict__ W, long ldw,
                const float* __restrict__ bias, float* __restrict__ Cf, long ldc,
                __bf16* __restrict__ Cb, long ldcb, int Nc, int K, int flags) {
  __shared__ __bf16 smem[2][2][128 * 64];  // [buf][A/W][row*64+k] : 64 KB
  int nbn = Nc >> 7;
  int bm = blockIdx.x / nbn, bn = blockIdx.x % nbn;
  int wave = threadIdx.x >> 5;
  int lane = threadIdx.x & 31;
  int wm = (wave & 3) * 32;   // wave's M offset inside macro-tile
  int wn = (wave >> 2) * 64;  // wave's N offset inside macro-tile
  int r = lane & 15;
  int kh = (lane & 16) ? 8 : 0;
  int m0 = (lane & 16) ? 8 : 0;
  int nk = K >> 6;

  v8f acc[2][4];
#pragma unroll
  for (int i = 0; i < 2; ++i)
#pragma unroll
    for (int j = 0; j < 4; ++j)
      acc[i][j] = (v8f){0.f, 0.f, 0.f, 0.f, 0.f, 0.f, 0.f, 0.f};

#if HAVE_TDM
  auto panel_issue = [&](int kp, int buf) {
    const __bf16* ga = A + (long)bm * 128 * lda + (long)kp * 64;
    const __bf16* gw = W + (long)bn * 128 * ldw + (long)kp * 64;
    tdm_load_2d((unsigned)(uintptr_t)&smem[buf][0][0], ga, 64, 128, lda);
    tdm_load_2d((unsigned)(uintptr_t)&smem[buf][1][0], gw, 64, 128, ldw);
  };
#else
  auto panel_copy = [&](int kp, int buf) {
    for (int e = threadIdx.x; e < 2048; e += 256) {  // 2 panels * 1024 v8 chunks
      int which = e >> 10;
      int idx = e & 1023;
      int row = idx >> 3, kc = (idx & 7) * 8;
      const __bf16* src =
          which ? (W + (long)(bn * 128 + row) * ldw + (long)kp * 64 + kc)
                : (A + (long)(bm * 128 + row) * lda + (long)kp * 64 + kc);
      *(v8bf*)&smem[buf][which][row * 64 + kc] = *(const v8bf*)src;
    }
  };
#endif

  auto compute = [&](int buf) {
    const __bf16* As = &smem[buf][0][0];
    const __bf16* Ws = &smem[buf][1][0];
#pragma unroll
    for (int k2 = 0; k2 < 64; k2 += 32) {
      v16bf af0 = load_frag32(As + (wm + r) * 64 + k2 + kh);
      v16bf af1 = load_frag32(As + (wm + 16 + r) * 64 + k2 + kh);
      v16bf wf0 = load_frag32(Ws + (wn + r) * 64 + k2 + kh);
      v16bf wf1 = load_frag32(Ws + (wn + 16 + r) * 64 + k2 + kh);
      v16bf wf2 = load_frag32(Ws + (wn + 32 + r) * 64 + k2 + kh);
      v16bf wf3 = load_frag32(Ws + (wn + 48 + r) * 64 + k2 + kh);
      acc[0][0] = wmma_bf16(af0, wf0, acc[0][0]);
      acc[0][1] = wmma_bf16(af0, wf1, acc[0][1]);
      acc[0][2] = wmma_bf16(af0, wf2, acc[0][2]);
      acc[0][3] = wmma_bf16(af0, wf3, acc[0][3]);
      acc[1][0] = wmma_bf16(af1, wf0, acc[1][0]);
      acc[1][1] = wmma_bf16(af1, wf1, acc[1][1]);
      acc[1][2] = wmma_bf16(af1, wf2, acc[1][2]);
      acc[1][3] = wmma_bf16(af1, wf3, acc[1][3]);
    }
  };

#if HAVE_TDM
  if (wave == 0) panel_issue(0, 0);
  for (int kp = 0; kp < nk; ++kp) {
    int buf = kp & 1;
    if (wave == 0) __builtin_amdgcn_s_wait_tensorcnt(0);
    __syncthreads();  // panel kp resident; all waves done reading buf^1
    if (wave == 0 && kp + 1 < nk) panel_issue(kp + 1, buf ^ 1);
    compute(buf);
  }
#else
  panel_copy(0, 0);
  __syncthreads();
  for (int kp = 0; kp < nk; ++kp) {
    int buf = kp & 1;
    if (kp + 1 < nk) panel_copy(kp + 1, buf ^ 1);
    compute(buf);
    __syncthreads();
  }
#endif

#pragma unroll
  for (int i = 0; i < 2; ++i)
#pragma unroll
    for (int j = 0; j < 4; ++j) {
      int col = bn * 128 + wn + j * 16 + r;
      float bv = bias ? bias[col] : 0.f;
#pragma unroll
      for (int t = 0; t < 8; ++t) {
        long row = bm * 128 + wm + i * 16 + m0 + t;
        float v = acc[i][j][t] + bv;
        if (flags & FLAG_ACC) v += Cf[row * ldc + col];
        if (flags & FLAG_RELU) v = fmaxf(v, 0.f);
        if (Cf) Cf[row * ldc + col] = v;
        if (Cb) Cb[row * ldcb + col] = (__bf16)v;
      }
    }
}

// --------------------------- pair-bias precompute --------------------------
__global__ void __launch_bounds__(256)
bias_kernel(const __bf16* __restrict__ z, const float* __restrict__ Wb,
            const float* __restrict__ bb, float* __restrict__ biasb) {
  __shared__ float wb[12 * 128];
  __shared__ float bbs[12];
  for (int t = threadIdx.x; t < 1536; t += 256) wb[t] = Wb[t];
  if (threadIdx.x < 12) bbs[threadIdx.x] = bb[threadIdx.x];
  __syncthreads();
  int ij = blockIdx.x * 256 + threadIdx.x;  // 384*384 exact
  int i = ij / 384, j = ij % 384;
  const __bf16* zp = z + (long)ij * 128;
  float acc[12];
#pragma unroll
  for (int h = 0; h < 12; ++h) acc[h] = bbs[h];
  for (int c = 0; c < 128; ++c) {
    float zc = (float)zp[c];
#pragma unroll
    for (int h = 0; h < 12; ++h) acc[h] += zc * wb[h * 128 + c];
  }
  for (int h = 0; h < 12; ++h)
    biasb[((long)h * 384 + i) * 384 + j] = acc[h];
}

// ------------------- v / v_pts rearrange (per block iter) ------------------
__global__ void rearrange_v_kernel(const float* __restrict__ kv,
                                   const float* __restrict__ kvp,
                                   __bf16* __restrict__ vT,
                                   __bf16* __restrict__ vpT) {
  int idx = blockIdx.x * 256 + threadIdx.x;  // 73728 + 147456 exact
  if (idx < 73728) {
    int j = idx % 384;
    int c = (idx / 384) & 15;
    int h = idx / 6144;
    vT[idx] = (__bf16)kv[(long)j * 384 + h * 32 + 16 + c];
  } else {
    int k = idx - 73728;
    int j = k % 384;
    int col = (k / 384) & 31;
    int h = k / 12288;
    float v = 0.f;
    if (col < 24) {
      int d = col >> 3, p = col & 7;
      v = kvp[(long)j * 432 + d * 144 + h * 12 + 4 + p];
    }
    vpT[k] = (__bf16)v;
  }
}

// --------------------------- attention + softmax ---------------------------
__global__ void __launch_bounds__(256)
attn_kernel(const float* __restrict__ q, const float* __restrict__ kv,
            const float* __restrict__ qp, const float* __restrict__ kvp,
            const float* __restrict__ biasb, const float* __restrict__ head_w,
            const float* __restrict__ mask, __bf16* __restrict__ a) {
  int wave = threadIdx.x >> 5;
  int lane = threadIdx.x & 31;
  int id = blockIdx.x * 8 + wave;  // 12*384 exact
  int h = id / 384, i = id % 384;
  float qv[16];
#pragma unroll
  for (int c = 0; c < 16; ++c) qv[c] = q[(long)i * 192 + h * 16 + c];
  float qpt[4][3];
#pragma unroll
  for (int p = 0; p < 4; ++p)
#pragma unroll
    for (int d = 0; d < 3; ++d)
      qpt[p][d] = qp[(long)i * 144 + d * 48 + h * 4 + p];
  float hw = log1pf(__expf(head_w[h])) * 0.13608277f;  // softplus * sqrt(1/54)
  float mi = mask[i];
  const float* brow = biasb + ((long)h * 384 + i) * 384;
  float lg[12];
  float mx = -1e30f;
#pragma unroll
  for (int t = 0; t < 12; ++t) {
    int j = lane + t * 32;
    const float* kj = kv + (long)j * 384 + h * 32;  // k half
    float sd = 0.f;
#pragma unroll
    for (int c = 0; c < 16; ++c) sd += qv[c] * kj[c];
    float pt = 0.f;
    const float* kp = kvp + (long)j * 432 + h * 12;
#pragma unroll
    for (int p = 0; p < 4; ++p)
#pragma unroll
      for (int d = 0; d < 3; ++d) {
        float dd = qpt[p][d] - kp[d * 144 + p];
        pt += dd * dd;
      }
    float l = sd * 0.14433757f + 0.57735027f * brow[j] - 0.5f * hw * pt +
              1e5f * (mi * mask[j] - 1.f);
    lg[t] = l;
    mx = fmaxf(mx, l);
  }
#pragma unroll
  for (int off = 16; off; off >>= 1) mx = fmaxf(mx, __shfl_xor(mx, off, 32));
  float sum = 0.f;
#pragma unroll
  for (int t = 0; t < 12; ++t) { lg[t] = __expf(lg[t] - mx); sum += lg[t]; }
#pragma unroll
  for (int off = 16; off; off >>= 1) sum += __shfl_xor(sum, off, 32);
  float inv = 1.f / sum;
  __bf16* arow = a + ((long)i * 16 + h) * 384;
#pragma unroll
  for (int t = 0; t < 12; ++t) arow[lane + t * 32] = (__bf16)(lg[t] * inv);
}

// --------------- o = a@v and o_pt = a@v_pts (per-head WMMA) ----------------
__global__ void __launch_bounds__(256)
o_opt_kernel(const __bf16* __restrict__ a, const __bf16* __restrict__ vT,
             const __bf16* __restrict__ vpT, float* __restrict__ o_f,
             float* __restrict__ opt_f) {
  int wave = threadIdx.x >> 5;
  int lane = threadIdx.x & 31;
  int id = blockIdx.x * 8 + wave;  // 288 + 576 = 864 exact
  int r = lane & 15;
  int kh = (lane & 16) ? 8 : 0;
  int h, mt, nt = 0;
  const __bf16* Wp;
  bool is_o = (id < 288);
  if (is_o) {
    h = id / 24; mt = id % 24;
    Wp = vT + (long)(h * 16 + r) * 384 + kh;
  } else {
    int id2 = id - 288;
    h = id2 / 48;
    int rem = id2 % 48;
    mt = rem >> 1; nt = rem & 1;
    Wp = vpT + (long)(h * 32 + nt * 16 + r) * 384 + kh;
  }
  const __bf16* Ap = a + ((long)(mt * 16 + r) * 16 + h) * 384 + kh;
  v8f acc = {0.f, 0.f, 0.f, 0.f, 0.f, 0.f, 0.f, 0.f};
  for (int k = 0; k < 384; k += 32)
    acc = wmma_bf16(load_frag32(Ap + k), load_frag32(Wp + k), acc);
  int m0 = (lane & 16) ? 8 : 0;
#pragma unroll
  for (int t = 0; t < 8; ++t) {
    long row = mt * 16 + m0 + t;
    if (is_o) o_f[row * 192 + h * 16 + r] = acc[t];
    else      opt_f[row * 384 + h * 32 + nt * 16 + r] = acc[t];
  }
}

// -------- o_pair[i] = a_heads(16 x 384) @ z[i](384 x 128) via WMMA ---------
__global__ void __launch_bounds__(256)
opair_kernel(const __bf16* __restrict__ a, const __bf16* __restrict__ z,
             float* __restrict__ opair_f) {
  int wave = threadIdx.x >> 5;
  int lane = threadIdx.x & 31;
  int id = blockIdx.x * 8 + wave;  // 384*8 exact
  int i = id >> 3, ct = id & 7;
  int r = lane & 15;
  int kh = (lane & 16) ? 8 : 0;
  const __bf16* Ap = a + ((long)i * 16 + r) * 384 + kh;
  const __bf16* Zp = z + (long)i * 49152 + ct * 16 + r;
  v8f acc = {0.f, 0.f, 0.f, 0.f, 0.f, 0.f, 0.f, 0.f};
  for (int k = 0; k < 384; k += 32) {
    v16bf af = load_frag32(Ap + k);
    v16bf bf;
#pragma unroll
    for (int t = 0; t < 8; ++t) {
      bf[t]     = Zp[(long)(k + kh + t) * 128];
      bf[8 + t] = Zp[(long)(k + kh + 16 + t) * 128];
    }
    acc = wmma_bf16(af, bf, acc);
  }
  int col = ct * 16 + r;
  int m0 = (lane & 16) ? 8 : 0;
#pragma unroll
  for (int t = 0; t < 8; ++t) {
    int h = m0 + t;
    if (h < 12) opair_f[(long)i * 1536 + h * 128 + col] = acc[t];
  }
}

// ------------------- concat [o | x | y | z | norm | o_pair] ----------------
__global__ void __launch_bounds__(256)
finalize_kernel(const float* __restrict__ o_f, const float* __restrict__ opt_f,
                const float* __restrict__ opair_f, __bf16* __restrict__ cat) {
  long i = blockIdx.x;
  for (int c = threadIdx.x; c < 2112; c += 256) {
    float v;
    if (c < 192) {
      v = o_f[i * 192 + c];
    } else if (c < 480) {
      int d = (c - 192) / 96;
      int idx = (c - 192) % 96;
      int h = idx >> 3, p = idx & 7;
      v = opt_f[i * 384 + h * 32 + d * 8 + p];
    } else if (c < 576) {
      int idx = c - 480;
      int h = idx >> 3, p = idx & 7;
      float x = opt_f[i * 384 + h * 32 + p];
      float y = opt_f[i * 384 + h * 32 + 8 + p];
      float zc = opt_f[i * 384 + h * 32 + 16 + p];
      v = sqrtf(x * x + y * y + zc * zc + 1e-8f);
    } else {
      v = opair_f[i * 1536 + (c - 576)];
    }
    cat[i * 2112 + c] = (__bf16)v;
  }
}

// ===========================================================================
extern "C" void kernel_launch(void* const* d_in, const int* in_sizes, int n_in,
                              void* d_out, int out_size, void* d_ws, size_t ws_size,
                              hipStream_t stream) {
  (void)in_sizes; (void)n_in; (void)ws_size;
  const float* s_in   = (const float*)d_in[0];
  const float* z_in   = (const float*)d_in[1];
  const float* mask   = (const float*)d_in[2];
  const float* ln_s_g = (const float*)d_in[3];
  const float* ln_s_b = (const float*)d_in[4];
  const float* ln_z_g = (const float*)d_in[5];
  const float* ln_z_b = (const float*)d_in[6];
  const float* W_in   = (const float*)d_in[7];
  const float* b_in   = (const float*)d_in[8];
  const float* Wq     = (const float*)d_in[9];
  const float* bq     = (const float*)d_in[10];
  const float* Wkv    = (const float*)d_in[11];
  const float* bkv    = (const float*)d_in[12];
  const float* Wqp    = (const float*)d_in[13];
  const float* bqp    = (const float*)d_in[14];
  const float* Wkvp   = (const float*)d_in[15];
  const float* bkvp   = (const float*)d_in[16];
  const float* Wb     = (const float*)d_in[17];
  const float* bb     = (const float*)d_in[18];
  const float* head_w = (const float*)d_in[19];
  const float* Wout   = (const float*)d_in[20];
  const float* bout   = (const float*)d_in[21];
  const float* ln_ig  = (const float*)d_in[22];
  const float* ln_ib  = (const float*)d_in[23];
  const float* W1     = (const float*)d_in[24];
  const float* b1     = (const float*)d_in[25];
  const float* W2     = (const float*)d_in[26];
  const float* b2     = (const float*)d_in[27];
  const float* W3     = (const float*)d_in[28];
  const float* b3     = (const float*)d_in[29];
  const float* ln_tg  = (const float*)d_in[30];
  const float* ln_tb  = (const float*)d_in[31];

  char* ws = (char*)d_ws;
  size_t off = 0;
  auto take = [&](size_t bytes) -> char* {
    char* p = ws + off;
    off = (off + bytes + 255) & ~(size_t)255;
    return p;
  };
  __bf16* zb      = (__bf16*)take(384UL * 384 * 128 * 2);
  float*  biasb   = (float*) take(12UL * 384 * 384 * 4);
  float*  S       = (float*) take(384UL * 384 * 4);
  __bf16* Sb      = (__bf16*)take(384UL * 384 * 2);
  __bf16* tmpb    = (__bf16*)take(384UL * 384 * 2);
  __bf16* Winb    = (__bf16*)take(147456UL * 2);
  __bf16* Wqb     = (__bf16*)take(73728UL * 2);
  __bf16* Wkvb    = (__bf16*)take(147456UL * 2);
  __bf16* Wqpb    = (__bf16*)take(55296UL * 2);
  __bf16* Wkvpb   = (__bf16*)take(165888UL * 2);
  __bf16* Woutb   = (__bf16*)take(811008UL * 2);
  __bf16* W1b     = (__bf16*)take(147456UL * 2);
  __bf16* W2b     = (__bf16*)take(147456UL * 2);
  __bf16* W3b     = (__bf16*)take(147456UL * 2);
  float*  qb      = (float*) take(384UL * 192 * 4);
  float*  kvb     = (float*) take(384UL * 384 * 4);
  float*  qpb     = (float*) take(384UL * 144 * 4);
  float*  kvpb    = (float*) take(384UL * 432 * 4);
  __bf16* ab      = (__bf16*)take(384UL * 16 * 384 * 2);
  __bf16* vT      = (__bf16*)take(12UL * 16 * 384 * 2);
  __bf16* vpT     = (__bf16*)take(12UL * 32 * 384 * 2);
  float*  o_f     = (float*) take(384UL * 192 * 4);
  float*  opt_f   = (float*) take(384UL * 384 * 4);
  float*  opair_f = (float*) take(384UL * 1536 * 4);
  __bf16* catb    = (__bf16*)take(384UL * 2112 * 2);
  __bf16* t1b     = (__bf16*)take(384UL * 384 * 2);
  __bf16* t2b     = (__bf16*)take(384UL * 384 * 2);

  auto cvt = [&](const float* src, __bf16* dst, int n) {
    cvt_kernel<<<(n + 255) / 256, 256, 0, stream>>>(src, dst, n);
  };
  auto gemm = [&](const __bf16* A, long lda, const __bf16* W, long ldw,
                  const float* bias, float* Cf, long ldc, __bf16* Cb, long ldcb,
                  int M, int Nc, int K, int flags) {
    int tiles = (M / 16) * (Nc / 16);
    gemm_bf16_kernel<<<(tiles + 7) / 8, 256, 0, stream>>>(
        A, lda, W, ldw, bias, Cf, ldc, Cb, ldcb, M, Nc, K, flags, tiles);
  };
  // TDM path for M,Nc multiples of 128 and K multiple of 64
  auto gemm_big = [&](const __bf16* A, long lda, const __bf16* W, long ldw,
                      const float* bias, float* Cf, long ldc, __bf16* Cb,
                      long ldcb, int M, int Nc, int K, int flags) {
    int blocks = (M / 128) * (Nc / 128);
    gemm_tdm_kernel<<<blocks, 256, 0, stream>>>(A, lda, W, ldw, bias, Cf, ldc,
                                                Cb, ldcb, Nc, K, flags);
  };

  // ---- one-time setup ----
  cvt(W_in, Winb, 147456);
  cvt(Wq, Wqb, 73728);
  cvt(Wkv, Wkvb, 147456);
  cvt(Wqp, Wqpb, 55296);
  cvt(Wkvp, Wkvpb, 165888);
  cvt(Wout, Woutb, 811008);
  cvt(W1, W1b, 147456);
  cvt(W2, W2b, 147456);
  cvt(W3, W3b, 147456);
  zero_pad_a_kernel<<<(589824 + 255) / 256, 256, 0, stream>>>(ab);

  ln_kernel<<<384 * 384, 256, 0, stream>>>(z_in, ln_z_g, ln_z_b, nullptr, zb, 128);
  bias_kernel<<<576, 256, 0, stream>>>(zb, Wb, bb, biasb);
  ln_kernel<<<384, 256, 0, stream>>>(s_in, ln_s_g, ln_s_b, nullptr, tmpb, 384);
  gemm_big(tmpb, 384, Winb, 384, b_in, S, 384, Sb, 384, 384, 384, 384, 0);

  // ---- 8 shared-weight IPA + transition blocks ----
  for (int blk = 0; blk < 8; ++blk) {
    gemm(Sb, 384, Wqb, 384, bq, qb, 192, nullptr, 0, 384, 192, 384, 0);
    gemm_big(Sb, 384, Wkvb, 384, bkv, kvb, 384, nullptr, 0, 384, 384, 384, 0);
    gemm(Sb, 384, Wqpb, 384, bqp, qpb, 144, nullptr, 0, 384, 144, 384, 0);
    gemm(Sb, 384, Wkvpb, 384, bkvp, kvpb, 432, nullptr, 0, 384, 432, 384, 0);
    rearrange_v_kernel<<<864, 256, 0, stream>>>(kvb, kvpb, vT, vpT);
    attn_kernel<<<576, 256, 0, stream>>>(qb, kvb, qpb, kvpb, biasb, head_w,
                                         mask, ab);
    o_opt_kernel<<<108, 256, 0, stream>>>(ab, vT, vpT, o_f, opt_f);
    opair_kernel<<<384, 256, 0, stream>>>(ab, zb, opair_f);
    finalize_kernel<<<384, 256, 0, stream>>>(o_f, opt_f, opair_f, catb);
    gemm_big(catb, 2112, Woutb, 2112, bout, S, 384, nullptr, 0, 384, 384, 2112,
             FLAG_ACC);
    ln_kernel<<<384, 256, 0, stream>>>(S, ln_ig, ln_ib, S, Sb, 384);
    gemm_big(Sb, 384, W1b, 384, b1, nullptr, 0, t1b, 384, 384, 384, 384,
             FLAG_RELU);
    gemm_big(t1b, 384, W2b, 384, b2, nullptr, 0, t2b, 384, 384, 384, 384,
             FLAG_RELU);
    gemm_big(t2b, 384, W3b, 384, b3, S, 384, nullptr, 0, 384, 384, 384,
             FLAG_ACC);
    ln_kernel<<<384, 256, 0, stream>>>(S, ln_tg, ln_tb, S, Sb, 384);
  }

  (void)hipMemcpyAsync(d_out, S, (size_t)out_size * sizeof(float),
                       hipMemcpyDeviceToDevice, stream);
}
